// lstmd_53377853555285
// MI455X (gfx1250) — matrix operand order repaired and checked by reading, not compiled, not run on hardware
//
#include <hip/hip_runtime.h>

// ---------------------------------------------------------------------------
// LSTM decoder for MI455X (gfx1250, wave32, WMMA f32_16x16x32_f16)
// B=32768, HIDDEN=256, IN=72, 8 steps. Compute-bound (~194 GFLOP) -> f16 WMMA.
// One workgroup per 64 batch rows; h/c/gates live in LDS across all 8 steps.
// Each wave owns an 8-tile N slice and ALL 4 M-subtiles: every weight fragment
// is loaded once per workgroup per step and reused across 4 WMMAs in registers
// (cuts L2 weight traffic 4x vs m-replicated streaming).
// ---------------------------------------------------------------------------

typedef __attribute__((ext_vector_type(16))) _Float16 v16h;
typedef __attribute__((ext_vector_type(8)))  _Float16 v8h;
typedef __attribute__((ext_vector_type(8)))  float    v8f;

#define HID   256
#define INW   72
#define KX    96      // x part of K, padded 72 -> 96
#define KCAT  352     // 96 + 256 combined K for the gate GEMM
#define NG    1024    // 4*HID gate columns
#define MT    64      // batch rows per workgroup (4 m-subtiles)
#define NSTEP 8

// workspace layout (bytes)
#define WS_WG   0           // f16 [1024][352]
#define WS_WH   720896      // f16 [256][256]
#define WS_WC   851968      // f16 [256][256]
#define WS_WO   983040      // f16 [80][256]
#define WS_BG   1024000     // f32 [1024]

__device__ __forceinline__ v16h cat16(v8h lo, v8h hi) {
  return __builtin_shufflevector(lo, hi, 0,1,2,3,4,5,6,7,8,9,10,11,12,13,14,15);
}

// A fragment (16x32 f16): lane L = row M=L&15; halves 0..15 <- K = kbase+16*lh+(0..15)
//   (two ds_load_b128)
__device__ __forceinline__ v16h ld_a_frag(const _Float16* row, int kbase, int lh) {
  const v8h* q0 = (const v8h*)(row + kbase + 16 * lh);
  const v8h* q1 = (const v8h*)(row + kbase + 16 * lh + 8);
  return cat16(*q0, *q1);
}

// B fragment (32x16 f16): lane L = col N=L&15; 16 consecutive K halves starting
// at kbase + 16*lh, read directly from row-major W[n][k] (two global_load_b128)
__device__ __forceinline__ v16h ld_b_frag(const _Float16* W, int n, int Ks,
                                          int kbase, int lh) {
  const _Float16* p = W + (size_t)n * Ks + kbase + 16 * lh;
  const v8h* q = (const v8h*)p;
  return cat16(q[0], q[1]);
}

__device__ __forceinline__ v8f wmma_f16(v16h a, v16h b, v8f c) {
  return __builtin_amdgcn_wmma_f32_16x16x32_f16(false, a, false, b,
                                                (short)0, c, false, false);
}

__device__ __forceinline__ float sigmoid_f(float x) {
  return 1.0f / (1.0f + __expf(-x));
}

// -------------------- weight convert / pack (runs once per launch) ----------
__global__ void lstm_pack(const float* __restrict__ W_ih, const float* __restrict__ W_hh,
                          const float* __restrict__ b_ih, const float* __restrict__ b_hh,
                          const float* __restrict__ Wh,   const float* __restrict__ Wc,
                          const float* __restrict__ Wo,
                          _Float16* __restrict__ Wg, _Float16* __restrict__ Wh16,
                          _Float16* __restrict__ Wc16, _Float16* __restrict__ Wo16,
                          float* __restrict__ bg) {
  int idx = blockIdx.x * blockDim.x + threadIdx.x;
  const int NWG = NG * KCAT;     // 360448
  const int NWH = HID * HID;     // 65536
  const int NWO = 80 * HID;      // 20480
  if (idx < NWG) {
    int n = idx / KCAT, k = idx % KCAT;
    float v = 0.0f;
    if (k < INW)      v = W_ih[n * INW + k];
    else if (k >= KX) v = W_hh[n * HID + (k - KX)];
    Wg[idx] = (_Float16)v;
  } else if ((idx -= NWG) < NWH) {
    Wh16[idx] = (_Float16)Wh[idx];
  } else if ((idx -= NWH) < NWH) {
    Wc16[idx] = (_Float16)Wc[idx];
  } else if ((idx -= NWH) < NWO) {
    int n = idx / HID;
    Wo16[idx] = (_Float16)((n < INW) ? Wo[idx] : 0.0f);
  } else if ((idx -= NWO) < NG) {
    bg[idx] = b_ih[idx] + b_hh[idx];
  }
}

// -------------------- persistent recurrent kernel ---------------------------
__global__ __launch_bounds__(256, 1)
void lstm_main(const float* __restrict__ z,
               const _Float16* __restrict__ Wg,   const float* __restrict__ bg,
               const _Float16* __restrict__ Wh16, const float* __restrict__ bh,
               const _Float16* __restrict__ Wc16, const float* __restrict__ bc,
               const _Float16* __restrict__ Wo16, const float* __restrict__ bo,
               float* __restrict__ out) {
  extern __shared__ char smem[];
  _Float16* Acat = (_Float16*)smem;                        // [MT][KCAT] f16 : [x|h]
  _Float16* G    = (_Float16*)(smem + MT * KCAT * 2);      // [MT][NG]   f16 : activated gates
  float*    Gf   = (float*)(smem + MT * KCAT * 2);         // init staging view [MT][512] f32
  float*    cst  = (float*)(smem + MT * KCAT * 2 + MT * NG * 2); // [MT][HID] f32 cell state

  const int tid  = threadIdx.x;
  const int lane = tid & 31;
  const int wave = tid >> 5;
  const int lh   = lane >> 4;   // lane half select
  const int lcol = lane & 15;   // N / M within subtile
  const int b0   = blockIdx.x * MT;

  // ---- init: zero the x region (covers padded cols 72..95 forever), load z0 as f16
  for (int i = tid; i < MT * KX; i += 256)
    Acat[(i / KX) * KCAT + (i % KX)] = (_Float16)0.0f;
  for (int r = 0; r < MT; ++r)
    Acat[r * KCAT + KX + tid] = (_Float16)z[(size_t)(b0 + r) * HID + tid];
  __syncthreads();

  // ---- init GEMMs: h = z0@Wh^T + bh , c = z0@Wc^T + bc   (staged f32 in Gf)
  {
    const int mat = wave >> 2;              // 0: Wh->h, 1: Wc->c
    const int jb  = (wave & 3) * 4;         // 4 n-subtiles per wave
    const _Float16* W    = mat ? Wc16 : Wh16;
    const float*    bias = mat ? bc : bh;
    v8f acc[4][4] = {};                     // [jj][m]
#pragma unroll
    for (int kc = 0; kc < 8; ++kc) {
      v16h am[4];
#pragma unroll
      for (int m = 0; m < 4; ++m)
        am[m] = ld_a_frag(Acat + (16 * m + lcol) * KCAT + KX, 32 * kc, lh);
#pragma unroll
      for (int jj = 0; jj < 4; ++jj) {
        v16h b = ld_b_frag(W, 16 * (jb + jj) + lcol, HID, 32 * kc, lh);
#pragma unroll
        for (int m = 0; m < 4; ++m)
          acc[jj][m] = wmma_f16(am[m], b, acc[jj][m]);
      }
    }
#pragma unroll
    for (int jj = 0; jj < 4; ++jj) {
      int n = 16 * (jb + jj) + lcol;
      float bv = bias[n];
      int col = n + (mat ? HID : 0);        // h -> cols 0..255, c -> 256..511
#pragma unroll
      for (int m = 0; m < 4; ++m)
#pragma unroll
        for (int r = 0; r < 8; ++r)
          Gf[(16 * m + r + 8 * lh) * 512 + col] = acc[jj][m][r] + bv;
    }
  }
  __syncthreads();
  for (int r = 0; r < MT; ++r) {            // staging -> state
    float hv = Gf[r * 512 + tid];
    float cv = Gf[r * 512 + HID + tid];
    Acat[r * KCAT + KX + tid] = (_Float16)hv;
    cst[r * HID + tid]        = cv;
  }
  __syncthreads();

  // ---- 8 recurrent steps
  for (int s = 0; s < NSTEP; ++s) {
    // (1) gates = [x|h] @ Wg^T + bg; each wave: 8 n-tiles x all 4 m-subtiles.
    //     B fragment loaded once, reused by 4 WMMAs (4x less L2 weight traffic).
    {
      const int gate = wave >> 1;           // 0:i 1:f 2:g 3:o (uniform per wave)
      const int jb0  = wave * 8;
#pragma unroll
      for (int jg = 0; jg < 2; ++jg) {      // two groups of 4 n-tiles
        const int jb = jb0 + 4 * jg;
        v8f acc[4][4] = {};                 // [jj][m]
#pragma unroll
        for (int kc = 0; kc < 11; ++kc) {
          v16h am[4];
#pragma unroll
          for (int m = 0; m < 4; ++m)
            am[m] = ld_a_frag(Acat + (16 * m + lcol) * KCAT, 32 * kc, lh);
#pragma unroll
          for (int jj = 0; jj < 4; ++jj) {
            v16h b = ld_b_frag(Wg, 16 * (jb + jj) + lcol, KCAT, 32 * kc, lh);
#pragma unroll
            for (int m = 0; m < 4; ++m)
              acc[jj][m] = wmma_f16(am[m], b, acc[jj][m]);
          }
        }
#pragma unroll
        for (int jj = 0; jj < 4; ++jj) {
          int n = 16 * (jb + jj) + lcol;
          float bv = bg[n];
#pragma unroll
          for (int m = 0; m < 4; ++m)
#pragma unroll
            for (int r = 0; r < 8; ++r) {
              float v = acc[jj][m][r] + bv;
              v = (gate == 2) ? tanhf(v) : sigmoid_f(v);
              G[(16 * m + r + 8 * lh) * NG + n] = (_Float16)v;  // activated, |v|<=1
            }
        }
      }
    }
    __syncthreads();

    // (2) elementwise LSTM update; h written back as f16 A-operand
    for (int r = 0; r < MT; ++r) {
      float iv = (float)G[r * NG + tid];
      float fv = (float)G[r * NG + 256 + tid];
      float gv = (float)G[r * NG + 512 + tid];
      float ov = (float)G[r * NG + 768 + tid];
      float cv = fv * cst[r * HID + tid] + iv * gv;
      cst[r * HID + tid] = cv;
      Acat[r * KCAT + KX + tid] = (_Float16)(ov * tanhf(cv));
    }
    __syncthreads();

    // (3) y = sigmoid(h @ Wo^T + bo): 5 n-tiles (padded 80), waves 0..4, all 4 m.
    if (wave < 5) {
      const int j = wave;
      v8f acc[4] = {};
#pragma unroll
      for (int kc = 0; kc < 8; ++kc) {
        v16h am[4];
#pragma unroll
        for (int m = 0; m < 4; ++m)
          am[m] = ld_a_frag(Acat + (16 * m + lcol) * KCAT + KX, 32 * kc, lh);
        v16h b = ld_b_frag(Wo16, 16 * j + lcol, HID, 32 * kc, lh);
#pragma unroll
        for (int m = 0; m < 4; ++m)
          acc[m] = wmma_f16(am[m], b, acc[m]);
      }
      int  col   = 16 * j + lcol;
      bool valid = col < INW;
      float bv = valid ? bo[col] : 0.0f;
#pragma unroll
      for (int m = 0; m < 4; ++m)
#pragma unroll
        for (int r = 0; r < 8; ++r) {
          float y = sigmoid_f(acc[m][r] + bv);
          int row = 16 * m + r + 8 * lh;
          if (valid) {
            out[((size_t)(b0 + row) * NSTEP + s) * INW + col] = y;
            Acat[row * KCAT + col] = (_Float16)y;   // feed back as next x
          }
        }
    }
    __syncthreads();
  }
}

// ---------------------------------------------------------------------------
extern "C" void kernel_launch(void* const* d_in, const int* in_sizes, int n_in,
                              void* d_out, int out_size, void* d_ws, size_t ws_size,
                              hipStream_t stream) {
  (void)n_in; (void)out_size; (void)ws_size;
  const float* z    = (const float*)d_in[0];
  const float* W_ih = (const float*)d_in[1];
  const float* W_hh = (const float*)d_in[2];
  const float* b_ih = (const float*)d_in[3];
  const float* b_hh = (const float*)d_in[4];
  const float* Wh   = (const float*)d_in[5];
  const float* bh   = (const float*)d_in[6];
  const float* Wc   = (const float*)d_in[7];
  const float* bc   = (const float*)d_in[8];
  const float* Wo   = (const float*)d_in[9];
  const float* bo   = (const float*)d_in[10];
  float* out = (float*)d_out;

  const int B = in_sizes[0] / HID;

  char* ws = (char*)d_ws;
  _Float16* Wg   = (_Float16*)(ws + WS_WG);
  _Float16* Wh16 = (_Float16*)(ws + WS_WH);
  _Float16* Wc16 = (_Float16*)(ws + WS_WC);
  _Float16* Wo16 = (_Float16*)(ws + WS_WO);
  float*    bg   = (float*)   (ws + WS_BG);

  // pack/convert weights to f16 (L2-resident afterwards)
  const int packN = NG * KCAT + 2 * HID * HID + 80 * HID + NG;   // 513024
  lstm_pack<<<(packN + 255) / 256, 256, 0, stream>>>(
      W_ih, W_hh, b_ih, b_hh, Wh, Wc, Wo, Wg, Wh16, Wc16, Wo16, bg);

  // 64 batch rows per workgroup; LDS: Acat 44KB + gates(f16) 128KB + c(f32) 64KB
  const size_t lds = (size_t)MT * KCAT * 2 + (size_t)MT * NG * 2 + (size_t)MT * HID * 4;
  lstm_main<<<B / MT, 256, lds, stream>>>(z, Wg, bg, Wh16, bh, Wc16, bc,
                                          Wo16, bo, out);
}
